// Sparse_attention_84146999263446
// MI455X (gfx1250) — compile-verified, matrix operand order checked by reading, not run on hardware
//
#include <hip/hip_runtime.h>
#include <math.h>

// ---------------------------------------------------------------------------
// Sparse token transformer block for MI455X (gfx1250, wave32, WMMA bf16).
// All heavy GEMMs use v_wmma_f32_16x16x32_bf16 (bf16 inputs, f32 accum).
// Softmax row reductions use DPP16 butterflies (no LDS crossbar traffic).
// Workspace requirement: ~522 MB carved from d_ws.
// ---------------------------------------------------------------------------

typedef __bf16 bf16_t;
typedef bf16_t bfv16 __attribute__((ext_vector_type(16)));
typedef float  fv8   __attribute__((ext_vector_type(8)));
typedef unsigned int uiv4 __attribute__((ext_vector_type(4)));

union FragU { bfv16 v; uiv4 u[2]; };

#define WMMA_BF16(A,B,C) \
  __builtin_amdgcn_wmma_f32_16x16x32_bf16(false,(A),false,(B),(short)0,(C),false,false)

// sizes
#define S_  30
#define P_  96
#define C_  1024
#define H_  16
#define HD_ 64
#define K_  6
#define MLP_ 4096
#define NSEL 576          // K_*P_
#define NT  17280         // S_*NSEL

__device__ inline unsigned short f2bf(float f){
  unsigned int u = __float_as_uint(f);
  u += 0x7FFFu + ((u >> 16) & 1u);           // round-to-nearest-even
  return (unsigned short)(u >> 16);
}
__device__ inline float bf2f(unsigned short h){
  return __uint_as_float(((unsigned int)h) << 16);
}
__device__ inline fv8 fv8_zero(){ fv8 z = {}; return z; }

// ---------------------------------------------------------------------------
// DPP16 lane exchange (pure VALU, row-local, no LDS).
//  0xB1  = quad_perm [1,0,3,2]  (xor 1)
//  0x4E  = quad_perm [2,3,0,1]  (xor 2)
//  0x141 = row_half_mirror      (xor-4 equivalent once quad-uniform)
//  0x128 = row_ror:8            (exact xor 8 within a 16-lane row)
// ---------------------------------------------------------------------------
template<int CTRL>
__device__ inline float dpp_xchg(float v){
  return __int_as_float(
      __builtin_amdgcn_update_dpp(0, __float_as_int(v), CTRL, 0xF, 0xF, true));
}
// reduce across the 16 lanes of each half-wave row; result in all 16 lanes
__device__ inline float row_max16(float v){
  v = fmaxf(v, dpp_xchg<0xB1>(v));
  v = fmaxf(v, dpp_xchg<0x4E>(v));
  v = fmaxf(v, dpp_xchg<0x141>(v));
  v = fmaxf(v, dpp_xchg<0x128>(v));
  return v;
}
__device__ inline float row_sum16(float v){
  v += dpp_xchg<0xB1>(v);
  v += dpp_xchg<0x4E>(v);
  v += dpp_xchg<0x141>(v);
  v += dpp_xchg<0x128>(v);
  return v;
}
// full wave32 sum: DPP within rows, one bpermute to cross rows
__device__ inline float wave_sum32(float v){
  v = row_sum16(v);
  v += __shfl_xor(v, 16, 32);
  return v;
}

// Load a 16x32 bf16 WMMA fragment (A layout; also B layout when the operand
// is stored [n][k] row-major with K contiguous).
// row must already include (lane&15); per-lane data = two 16B runs:
//   elems 0..7  -> K = (lane/16)*8 + 0..7
//   elems 8..15 -> K = 16 + (lane/16)*8 + 0..7
__device__ inline bfv16 frag_ld(const unsigned short* __restrict__ base,
                                int row, int ld, int k0, int lane){
  const unsigned short* p = base + (size_t)row * (size_t)ld + k0 + ((lane >> 4) << 3);
  FragU f;
  f.u[0] = *(const uiv4*)p;
  f.u[1] = *(const uiv4*)(p + 16);
  return f.v;
}

// ---------------------------------------------------------------------------
// f32 -> bf16 raw conversion (weights)
// ---------------------------------------------------------------------------
__global__ void cvt_bf16_kernel(const float* __restrict__ in,
                                unsigned short* __restrict__ out, int n){
  int i = blockIdx.x * 256 + threadIdx.x;
  if (i < n) out[i] = f2bf(in[i]);
}

// ---------------------------------------------------------------------------
// similarity + masked softmax + top-k  (1 wave per query row)
// ---------------------------------------------------------------------------
__global__ void topk_kernel(const float* __restrict__ tokens,
                            float* __restrict__ vals, int* __restrict__ idxs){
  int s = blockIdx.x;
  int lane = threadIdx.x;
  __shared__ float row[S_];
  const float* cs = tokens + (size_t)s * P_ * C_;   // cam[s] = tokens[s][0][:]
  for (int t = 0; t < S_; ++t){
    const float* ct = tokens + (size_t)t * P_ * C_;
    float acc = 0.f;
    for (int c = lane; c < C_; c += 32) acc += cs[c] * ct[c];
    acc = wave_sum32(acc);
    if (lane == 0) row[t] = acc * (1.0f / 32.0f);
  }
  if (lane == 0){
    int bs = s / 5;
    float p[S_]; float mx = -1e30f;
    for (int t = 0; t < S_; ++t){
      float v = row[t] - ((t / 5 == bs) ? 100000.0f : 0.0f);
      p[t] = v; mx = fmaxf(mx, v);
    }
    float sum = 0.f;
    for (int t = 0; t < S_; ++t){ p[t] = __expf(p[t] - mx); sum += p[t]; }
    float inv = 1.0f / sum;
    for (int t = 0; t < S_; ++t) p[t] *= inv;
    for (int kk = 0; kk < K_; ++kk){
      float best = -1.0f; int bi = 0;
      for (int t = 0; t < S_; ++t) if (p[t] > best){ best = p[t]; bi = t; }
      vals[s * K_ + kk] = best;
      idxs[s * K_ + kk] = bi;
      p[bi] = -2.0f;
    }
  }
}

// ---------------------------------------------------------------------------
// block-wide (256 thr) mean / rsqrt(var+eps) over 1024 elements
// ---------------------------------------------------------------------------
__device__ inline void block_stats_1024(float s1, float s2, float& mean, float& inv){
  s1 = wave_sum32(s1);
  s2 = wave_sum32(s2);
  __shared__ float a1[8], a2[8];
  int tid = threadIdx.x, wv = tid >> 5;
  if ((tid & 31) == 0){ a1[wv] = s1; a2[wv] = s2; }
  __syncthreads();
  float t1 = (tid < 8) ? a1[tid] : 0.f;
  float t2 = (tid < 8) ? a2[tid] : 0.f;
  #pragma unroll
  for (int m = 4; m >= 1; m >>= 1){ t1 += __shfl_xor(t1, m, 32); t2 += __shfl_xor(t2, m, 32); }
  if (tid == 0){ a1[0] = t1; a2[0] = t2; }
  __syncthreads();
  float S1 = a1[0], S2 = a2[0];
  mean = S1 * (1.0f / 1024.0f);
  float var = S2 * (1.0f / 1024.0f) - mean * mean;
  inv = rsqrtf(var + 1e-6f);
}

// gather selected tokens, LayerNorm1 -> x (f32 residual) + xb (bf16)
__global__ void gather_ln1_kernel(const float* __restrict__ tokens,
                                  const int* __restrict__ idxs,
                                  const float* __restrict__ w, const float* __restrict__ b,
                                  float* __restrict__ x, unsigned short* __restrict__ xb){
  int r = blockIdx.x, tid = threadIdx.x;
  int s = r / NSEL, rem = r % NSEL, kk = rem / P_, pp = rem % P_;
  int src = idxs[s * K_ + kk];
  const float* tp = tokens + ((size_t)src * P_ + pp) * C_;
  float v[4]; float s1 = 0.f, s2 = 0.f;
  #pragma unroll
  for (int i = 0; i < 4; ++i){ float t = tp[tid + i * 256]; v[i] = t; s1 += t; s2 += t * t; }
  float mean, inv;
  block_stats_1024(s1, s2, mean, inv);
  #pragma unroll
  for (int i = 0; i < 4; ++i){
    int c = tid + i * 256;
    x[(size_t)r * C_ + c]  = v[i];
    xb[(size_t)r * C_ + c] = f2bf((v[i] - mean) * inv * w[c] + b[c]);
  }
}

// LayerNorm2 over rows of x -> bf16
__global__ void ln_rows_kernel(const float* __restrict__ in,
                               const float* __restrict__ w, const float* __restrict__ b,
                               unsigned short* __restrict__ outb){
  int r = blockIdx.x, tid = threadIdx.x;
  const float* tp = in + (size_t)r * C_;
  float v[4]; float s1 = 0.f, s2 = 0.f;
  #pragma unroll
  for (int i = 0; i < 4; ++i){ float t = tp[tid + i * 256]; v[i] = t; s1 += t; s2 += t * t; }
  float mean, inv;
  block_stats_1024(s1, s2, mean, inv);
  #pragma unroll
  for (int i = 0; i < 4; ++i){
    int c = tid + i * 256;
    outb[(size_t)r * C_ + c] = f2bf((v[i] - mean) * inv * w[c] + b[c]);
  }
}

// ---------------------------------------------------------------------------
// WMMA bf16 GEMM: D[M,N] = A[M,K] * W[N,K]^T (+bias), templated epilogue.
// Block = 256 thr (8 waves), tile 64(M) x 128(N); wave = 32x32 (2x2 frags).
// EPI 0: outb = bf16(acc + bias)
// EPI 1: xio += ls * (acc + bias)          (residual, f32 in place)
// EPI 2: outb = bf16(gelu_exact(acc+bias))
// ---------------------------------------------------------------------------
template<int EPI>
__global__ void gemm_bf16_kernel(const unsigned short* __restrict__ A,
                                 const unsigned short* __restrict__ W,
                                 const float* __restrict__ bias,
                                 const float* __restrict__ ls,
                                 float* __restrict__ xio,
                                 unsigned short* __restrict__ outb,
                                 int M, int N, int Kd){
  int lane = threadIdx.x & 31;
  int wave = threadIdx.x >> 5;
  int wm = wave >> 2, wn = wave & 3;
  int m0 = blockIdx.x * 64  + wm * 32;
  int n0 = blockIdx.y * 128 + wn * 32;

  fv8 acc[2][2];
  #pragma unroll
  for (int i = 0; i < 2; ++i)
    #pragma unroll
    for (int j = 0; j < 2; ++j) acc[i][j] = fv8_zero();

  int rowA0 = m0 + (lane & 15), rowA1 = rowA0 + 16;
  int rowB0 = n0 + (lane & 15), rowB1 = rowB0 + 16;

  #pragma unroll 2
  for (int k0 = 0; k0 < Kd; k0 += 32){
    __builtin_prefetch(A + (size_t)rowA0 * Kd + k0 + 64, 0, 0);
    __builtin_prefetch(W + (size_t)rowB0 * Kd + k0 + 64, 0, 0);
    bfv16 a0 = frag_ld(A, rowA0, Kd, k0, lane);
    bfv16 a1 = frag_ld(A, rowA1, Kd, k0, lane);
    bfv16 b0 = frag_ld(W, rowB0, Kd, k0, lane);
    bfv16 b1 = frag_ld(W, rowB1, Kd, k0, lane);
    acc[0][0] = WMMA_BF16(a0, b0, acc[0][0]);
    acc[0][1] = WMMA_BF16(a0, b1, acc[0][1]);
    acc[1][0] = WMMA_BF16(a1, b0, acc[1][0]);
    acc[1][1] = WMMA_BF16(a1, b1, acc[1][1]);
  }

  int hl = lane >> 4, nn = lane & 15;
  #pragma unroll
  for (int i = 0; i < 2; ++i)
    #pragma unroll
    for (int j = 0; j < 2; ++j)
      #pragma unroll
      for (int e = 0; e < 8; ++e){
        int m = m0 + i * 16 + e + hl * 8;
        int n = n0 + j * 16 + nn;
        float v = acc[i][j][e] + bias[n];
        if (EPI == 0){
          outb[(size_t)m * N + n] = f2bf(v);
        } else if (EPI == 1){
          xio[(size_t)m * N + n] += ls[n] * v;
        } else {
          float g = 0.5f * v * (1.0f + erff(v * 0.70710678118654752f));
          outb[(size_t)m * N + n] = f2bf(g);
        }
      }
}

// ---------------------------------------------------------------------------
// per-head LayerNorm of q,k (HD=64) + split; v stored TRANSPOSED [r,h,d,key]
// so the P*V B-operand has K(=key) contiguous. 1 wave per (token, head).
// ---------------------------------------------------------------------------
__global__ void headln_split_kernel(const unsigned short* __restrict__ z,
                                    const float* __restrict__ qw, const float* __restrict__ qb,
                                    const float* __restrict__ kw, const float* __restrict__ kb,
                                    unsigned short* __restrict__ qn,
                                    unsigned short* __restrict__ kn,
                                    unsigned short* __restrict__ vt){
  int t = blockIdx.x, h = blockIdx.y, lane = threadIdx.x;
  size_t zb0 = (size_t)t * 3072 + h * HD_;
  float q0 = bf2f(z[zb0 + lane]),        q1 = bf2f(z[zb0 + lane + 32]);
  float k0 = bf2f(z[zb0 + 1024 + lane]), k1 = bf2f(z[zb0 + 1024 + lane + 32]);

  float qs1 = wave_sum32(q0 + q1);
  float qs2 = wave_sum32(q0 * q0 + q1 * q1);
  float ks1 = wave_sum32(k0 + k1);
  float ks2 = wave_sum32(k0 * k0 + k1 * k1);

  float qm = qs1 * (1.0f / 64.0f), qv = qs2 * (1.0f / 64.0f) - qm * qm;
  float km = ks1 * (1.0f / 64.0f), kv = ks2 * (1.0f / 64.0f) - km * km;
  float qi = rsqrtf(qv + 1e-6f), ki = rsqrtf(kv + 1e-6f);

  size_t ob0 = (size_t)t * C_ + h * HD_;
  qn[ob0 + lane]      = f2bf((q0 - qm) * qi * qw[lane]      + qb[lane]);
  qn[ob0 + lane + 32] = f2bf((q1 - qm) * qi * qw[lane + 32] + qb[lane + 32]);
  kn[ob0 + lane]      = f2bf((k0 - km) * ki * kw[lane]      + kb[lane]);
  kn[ob0 + lane + 32] = f2bf((k1 - km) * ki * kw[lane + 32] + kb[lane + 32]);

  int r = t / NSEL, key = t % NSEL;
  size_t vbase = ((size_t)(r * H_ + h)) * HD_;
  vt[(vbase + lane)      * NSEL + key] = z[zb0 + 2048 + lane];
  vt[(vbase + lane + 32) * NSEL + key] = z[zb0 + 2048 + lane + 32];
}

// ---------------------------------------------------------------------------
// Flash attention: 1 wave per 16-query tile, key tiles of 32, per (r,h).
// Scores: 4 WMMAs; softmax via DPP16 row reductions (pure VALU);
// probabilities bounced through per-wave LDS (ds stores -> s_wait_dscnt ->
// ds_load_b128) to re-layout into an A fragment; P*V: 4 WMMAs over d-slices.
// ---------------------------------------------------------------------------
__global__ void attn_kernel(const unsigned short* __restrict__ qn,
                            const unsigned short* __restrict__ kn,
                            const unsigned short* __restrict__ vt,
                            unsigned short* __restrict__ ob){
  int lane = threadIdx.x & 31;
  int wv   = threadIdx.x >> 5;
  int qt = blockIdx.x * 4 + wv;          // 0..35
  int h  = blockIdx.y;
  int r  = blockIdx.z;

  __shared__ __align__(16) unsigned short pl[4][16 * 32];
  unsigned short* myp = pl[wv];

  int q0 = qt * 16;
  int hl = lane >> 4, nn = lane & 15;
  int rowq = r * NSEL + q0 + nn;
  int col0 = h * HD_;

  bfv16 qf0 = frag_ld(qn, rowq, C_, col0,      lane);
  bfv16 qf1 = frag_ld(qn, rowq, C_, col0 + 32, lane);

  fv8 oacc[4];
  #pragma unroll
  for (int d = 0; d < 4; ++d) oacc[d] = fv8_zero();
  float rm[8], rs[8];
  #pragma unroll
  for (int e = 0; e < 8; ++e){ rm[e] = -1e30f; rs[e] = 0.f; }

  for (int kt = 0; kt < 18; ++kt){
    int k0k = kt * 32;
    int rowk0 = r * NSEL + k0k + nn;
    int rowk1 = rowk0 + 16;
    bfv16 kf00 = frag_ld(kn, rowk0, C_, col0,      lane);
    bfv16 kf01 = frag_ld(kn, rowk0, C_, col0 + 32, lane);
    bfv16 kf10 = frag_ld(kn, rowk1, C_, col0,      lane);
    bfv16 kf11 = frag_ld(kn, rowk1, C_, col0 + 32, lane);

    fv8 s0 = fv8_zero(), s1 = fv8_zero();
    s0 = WMMA_BF16(qf0, kf00, s0);
    s0 = WMMA_BF16(qf1, kf01, s0);
    s1 = WMMA_BF16(qf0, kf10, s1);
    s1 = WMMA_BF16(qf1, kf11, s1);

    float p0[8], p1[8], corr[8];
    #pragma unroll
    for (int e = 0; e < 8; ++e){
      float a = s0[e] * 0.125f;            // 1/sqrt(64)
      float b = s1[e] * 0.125f;
      float mt = row_max16(fmaxf(a, b));   // DPP16 butterfly, all 16 lanes
      float nm = fmaxf(rm[e], mt);
      float c  = __expf(rm[e] - nm);
      rm[e] = nm; corr[e] = c;
      float e0 = __expf(a - nm), e1 = __expf(b - nm);
      p0[e] = e0; p1[e] = e1;
      float ts = row_sum16(e0 + e1);       // DPP16 butterfly
      rs[e] = rs[e] * c + ts;
    }
    #pragma unroll
    for (int d = 0; d < 4; ++d)
      #pragma unroll
      for (int e = 0; e < 8; ++e) oacc[d][e] *= corr[e];

    // p -> LDS ([m][n], row stride 32), then reload as A fragment
    #pragma unroll
    for (int e = 0; e < 8; ++e){
      int m = e + 8 * hl;
      myp[m * 32 + nn]      = f2bf(p0[e]);
      myp[m * 32 + nn + 16] = f2bf(p1[e]);
    }
    asm volatile("s_wait_dscnt 0x0" ::: "memory");
    const unsigned short* pp = myp + nn * 32 + (hl << 3);
    FragU pf;
    pf.u[0] = *(const uiv4*)pp;
    pf.u[1] = *(const uiv4*)(pp + 16);

    #pragma unroll
    for (int d = 0; d < 4; ++d){
      int rowv = (r * H_ + h) * HD_ + d * 16 + nn;
      bfv16 vf = frag_ld(vt, rowv, NSEL, k0k, lane);
      oacc[d] = WMMA_BF16(pf.v, vf, oacc[d]);
    }
  }

  #pragma unroll
  for (int e = 0; e < 8; ++e){
    float inv = 1.0f / rs[e];
    int m = e + 8 * hl;
    size_t tok = (size_t)(r * NSEL + q0 + m);
    #pragma unroll
    for (int d = 0; d < 4; ++d)
      ob[tok * C_ + h * HD_ + d * 16 + nn] = f2bf(oacc[d][e] * inv);
  }
}

// ---------------------------------------------------------------------------
// weighted mean over the K selected slots -> output
// ---------------------------------------------------------------------------
__global__ void out_kernel(const float* __restrict__ x,
                           const float* __restrict__ vals,
                           float* __restrict__ out){
  int sp = blockIdx.x;                 // s * 96 + p
  int s = sp / P_, pp = sp % P_;
  int tid = threadIdx.x;
  #pragma unroll
  for (int i = 0; i < 4; ++i){
    int c = tid + i * 256;
    float acc = 0.f;
    #pragma unroll
    for (int kk = 0; kk < K_; ++kk)
      acc += vals[s * K_ + kk] * x[(((size_t)s * K_ + kk) * P_ + pp) * C_ + c];
    out[(size_t)sp * C_ + c] = acc * (1.0f / (float)K_);
  }
}

// ---------------------------------------------------------------------------
extern "C" void kernel_launch(void* const* d_in, const int* in_sizes, int n_in,
                              void* d_out, int out_size, void* d_ws, size_t ws_size,
                              hipStream_t stream){
  const float* tokens = (const float*)d_in[0];
  /* d_in[1] = idx (unused by forward) */
  const float* qkv_w  = (const float*)d_in[2];
  const float* qkv_b  = (const float*)d_in[3];
  const float* qnw    = (const float*)d_in[4];
  const float* qnb    = (const float*)d_in[5];
  const float* knw    = (const float*)d_in[6];
  const float* knb    = (const float*)d_in[7];
  const float* proj_w = (const float*)d_in[8];
  const float* proj_b = (const float*)d_in[9];
  const float* ls1    = (const float*)d_in[10];
  const float* n1w    = (const float*)d_in[11];
  const float* n1b    = (const float*)d_in[12];
  const float* n2w    = (const float*)d_in[13];
  const float* n2b    = (const float*)d_in[14];
  const float* fc1_w  = (const float*)d_in[15];
  const float* fc1_b  = (const float*)d_in[16];
  const float* fc2_w  = (const float*)d_in[17];
  const float* fc2_b  = (const float*)d_in[18];
  const float* ls2    = (const float*)d_in[19];

  // workspace carve (256B aligned)
  char* wsp = (char*)d_ws;
  auto carve = [&](size_t bytes) -> void* {
    void* p = (void*)wsp;
    wsp += (bytes + 255) & ~(size_t)255;
    return p;
  };
  unsigned short* wq   = (unsigned short*)carve((size_t)3072 * 1024 * 2);
  unsigned short* wp   = (unsigned short*)carve((size_t)1024 * 1024 * 2);
  unsigned short* w1   = (unsigned short*)carve((size_t)4096 * 1024 * 2);
  unsigned short* w2   = (unsigned short*)carve((size_t)1024 * 4096 * 2);
  float*          vals = (float*)carve((size_t)S_ * K_ * 4);
  int*            idxs = (int*)carve((size_t)S_ * K_ * 4);
  float*          x    = (float*)carve((size_t)NT * C_ * 4);
  unsigned short* xb   = (unsigned short*)carve((size_t)NT * C_ * 2);   // LN1 out, reused for LN2 out
  unsigned short* zb   = (unsigned short*)carve((size_t)NT * 3072 * 2); // qkv out
  unsigned short* qbuf = (unsigned short*)carve((size_t)NT * C_ * 2);
  unsigned short* kbuf = (unsigned short*)carve((size_t)NT * C_ * 2);
  unsigned short* vtb  = (unsigned short*)carve((size_t)NT * C_ * 2);   // v transposed [r,h,d,key]
  unsigned short* obuf = (unsigned short*)carve((size_t)NT * C_ * 2);
  unsigned short* mlph = (unsigned short*)carve((size_t)NT * MLP_ * 2);

  // weights -> bf16
  cvt_bf16_kernel<<<(3072 * 1024 + 255) / 256, 256, 0, stream>>>(qkv_w, wq, 3072 * 1024);
  cvt_bf16_kernel<<<(1024 * 1024 + 255) / 256, 256, 0, stream>>>(proj_w, wp, 1024 * 1024);
  cvt_bf16_kernel<<<(4096 * 1024 + 255) / 256, 256, 0, stream>>>(fc1_w, w1, 4096 * 1024);
  cvt_bf16_kernel<<<(4096 * 1024 + 255) / 256, 256, 0, stream>>>(fc2_w, w2, 4096 * 1024);

  // similarity / softmax / top-k
  topk_kernel<<<S_, 32, 0, stream>>>(tokens, vals, idxs);

  // gather + LN1
  gather_ln1_kernel<<<NT, 256, 0, stream>>>(tokens, idxs, n1w, n1b, x, xb);

  // QKV GEMM: [17280,1024] x [3072,1024]^T -> bf16 z
  gemm_bf16_kernel<0><<<dim3(NT / 64, 3072 / 128), 256, 0, stream>>>(
      xb, wq, qkv_b, nullptr, nullptr, zb, NT, 3072, 1024);

  // per-head LN + split (v transposed)
  headln_split_kernel<<<dim3(NT, H_), 32, 0, stream>>>(
      zb, qnw, qnb, knw, knb, qbuf, kbuf, vtb);

  // flash attention: grid (9 query-blocks, 16 heads, 30 rows), 4 waves/block
  attn_kernel<<<dim3(9, H_, S_), 128, 0, stream>>>(qbuf, kbuf, vtb, obuf);

  // proj GEMM + residual: x += ls1 * (o @ proj^T + b)
  gemm_bf16_kernel<1><<<dim3(NT / 64, 1024 / 128), 256, 0, stream>>>(
      obuf, wp, proj_b, ls1, x, nullptr, NT, 1024, 1024);

  // LN2
  ln_rows_kernel<<<NT, 256, 0, stream>>>(x, n2w, n2b, xb);

  // fc1 GEMM + exact GELU -> bf16 hidden
  gemm_bf16_kernel<2><<<dim3(NT / 64, 4096 / 128), 256, 0, stream>>>(
      xb, w1, fc1_b, nullptr, nullptr, mlph, NT, 4096, 1024);

  // fc2 GEMM + residual: x += ls2 * (h @ fc2^T + b)
  gemm_bf16_kernel<1><<<dim3(NT / 64, 1024 / 128), 256, 0, stream>>>(
      mlph, w2, fc2_b, ls2, x, nullptr, NT, 1024, 4096);

  // weighted mean over K slots -> out (1,30,96,1024)
  out_kernel<<<S_ * P_, 256, 0, stream>>>(x, vals, (float*)d_out);
}
